// MLPPredictor_81217831568089
// MI455X (gfx1250) — matrix operand order repaired and checked by reading, not compiled, not run on hardware
//
#include <hip/hip_runtime.h>

typedef __attribute__((ext_vector_type(16))) _Float16 v16h;
typedef __attribute__((ext_vector_type(8)))  _Float16 v8h;
typedef __attribute__((ext_vector_type(4)))  _Float16 v4h;
typedef __attribute__((ext_vector_type(8)))  float    v8f;
typedef __attribute__((ext_vector_type(4)))  float    v4f;

#define HFEAT 64      // per-node feature width
#define KDIM  128     // concat width (2*HFEAT)
#define TILES_PER_WAVE 4   // 64 edges per wave: amortizes B-register preload

// ---------------------------------------------------------------------------
// Pass 1: convert node features f32 -> f16 (node table then lives L2-resident
// at 12.8MB and every edge gather is half the bytes).
// ---------------------------------------------------------------------------
__global__ __launch_bounds__(256)
void cvt_f32_to_f16(const float* __restrict__ in, _Float16* __restrict__ out, int n4) {
    int i = blockIdx.x * blockDim.x + threadIdx.x;
    if (i < n4) {
        v4f f = ((const v4f*)in)[i];
        v4h o;
        o[0] = (_Float16)f[0]; o[1] = (_Float16)f[1];
        o[2] = (_Float16)f[2]; o[3] = (_Float16)f[3];
        ((v4h*)out)[i] = o;
    }
}

// ---------------------------------------------------------------------------
// A-fragment gather for one 16-edge tile.
// A is 16x32 f16: lane L owns edge row L&15; lanes 0-15 hold K {0..7,16..23},
// lanes 16-31 hold K {8..15,24..31} -> two contiguous 16B chunks per kstep
// (compiler merges the 4 ksteps per node into 4x global_load_b128).
// ---------------------------------------------------------------------------
template<bool STAGED>
__device__ __forceinline__ void load_A_tile(const void* __restrict__ h_ptr,
                                            const int* __restrict__ src,
                                            const int* __restrict__ dst,
                                            int tile, int row, int hi, int E,
                                            v16h (&A)[4]) {
    int e = tile * 16 + row;
    if (e >= E) e = E - 1;                     // safety clamp (E%16==0 here)
    const size_t sn = (size_t)src[e];
    const size_t dn = (size_t)dst[e];
    if (STAGED) {
        const _Float16* hf = (const _Float16*)h_ptr;
        #pragma unroll
        for (int ks = 0; ks < 4; ++ks) {
            const size_t node = (ks < 2) ? sn : dn;     // K 0..63 = src, 64..127 = dst
            const _Float16* p = hf + node * HFEAT + (ks & 1) * 32 + hi * 8;
            v8h lo  = *(const v8h*)(p);
            v8h hi8 = *(const v8h*)(p + 16);
            #pragma unroll
            for (int i = 0; i < 8; ++i) { A[ks][i] = lo[i]; A[ks][i + 8] = hi8[i]; }
        }
    } else {
        const float* hf = (const float*)h_ptr;
        #pragma unroll
        for (int ks = 0; ks < 4; ++ks) {
            const size_t node = (ks < 2) ? sn : dn;
            const float* p = hf + node * HFEAT + (ks & 1) * 32 + hi * 8;
            v4f f0 = *(const v4f*)(p);
            v4f f1 = *(const v4f*)(p + 4);
            v4f f2 = *(const v4f*)(p + 16);
            v4f f3 = *(const v4f*)(p + 20);
            #pragma unroll
            for (int i = 0; i < 4; ++i) {
                A[ks][i]      = (_Float16)f0[i];
                A[ks][i + 4]  = (_Float16)f1[i];
                A[ks][i + 8]  = (_Float16)f2[i];
                A[ks][i + 12] = (_Float16)f3[i];
            }
        }
    }
}

// ---------------------------------------------------------------------------
// Main kernel: each wave32 processes TILES_PER_WAVE tiles of 16 edges.
//   hidden[16x64] = relu( concat(h[src],h[dst])[16x128] @ W1^T + b1 )
//   score[16]     = hidden @ W2^T + b2
// W1 staged LDS once per block; all 16 B fragments hoisted to registers once
// per wave (single ds clause + wait); A gather double-buffered across tiles.
// ---------------------------------------------------------------------------
template<bool STAGED>
__global__ __launch_bounds__(256)
void edge_mlp_wmma(const void* __restrict__ h_ptr,
                   const int*  __restrict__ src,
                   const int*  __restrict__ dst,
                   const float* __restrict__ W1w,   // [64][128] row-major (out,in)
                   const float* __restrict__ W1b,   // [64]
                   const float* __restrict__ W2w,   // [64]
                   const float* __restrict__ W2b,   // [1]
                   float* __restrict__ out,         // [E]
                   int nTiles, int E)
{
    // Stage W1 as f16 in LDS: w1s[n][k], n=hidden feat, k=concat dim.
    __shared__ _Float16 w1s[HFEAT * KDIM];          // 16 KB of 320 KB/WGP
    const int tid = threadIdx.x;
    for (int i = tid; i < HFEAT * KDIM; i += 256)
        w1s[i] = (_Float16)W1w[i];
    __syncthreads();

    const int lane = tid & 31;
    const int row  = lane & 15;                     // edge row within tile (A-matrix M)
    const int hi   = lane >> 4;                     // half-wave -> K sub-block

    const int waveGlobal = blockIdx.x * 8 + (tid >> 5);
    const int t0 = waveGlobal * TILES_PER_WAVE;
    if (t0 >= nTiles) return;                       // wave-uniform: EXEC stays all-1s
    const int tEnd = (t0 + TILES_PER_WAVE < nTiles) ? (t0 + TILES_PER_WAVE) : nTiles;

    // ---- Hoist all 16 B fragments (W1^T) to registers: B[k][n] = W1[n][k].
    // Lane L holds col n = nt*16+(L&15), K = ks*32 + {hi*8 + 0..7, 16 + hi*8 + 0..7}
    // -> two 8-half chunks (16B aligned) per fragment; one ds clause total.
    v16h B[4][4];
    {
        const _Float16* bp = &w1s[row * KDIM + hi * 8];
        #pragma unroll
        for (int nt = 0; nt < 4; ++nt) {
            #pragma unroll
            for (int ks = 0; ks < 4; ++ks) {
                const _Float16* p = bp + nt * 16 * KDIM + ks * 32;
                v8h b0 = *(const v8h*)(p);
                v8h b1 = *(const v8h*)(p + 16);
                #pragma unroll
                for (int i = 0; i < 8; ++i) { B[nt][ks][i] = b0[i]; B[nt][ks][i + 8] = b1[i]; }
            }
        }
    }

    // Per-lane layer-2 constants: n = nt*16 + row.
    float b1n[4], w2n[4];
    #pragma unroll
    for (int nt = 0; nt < 4; ++nt) {
        b1n[nt] = W1b[nt * 16 + row];
        w2n[nt] = W2w[nt * 16 + row];
    }
    const float b2 = W2b[0];

    // ---- Software-pipelined tile loop: gather for t+1 overlaps WMMAs for t.
    v16h A[4], An[4];
    load_A_tile<STAGED>(h_ptr, src, dst, t0, row, hi, E, A);
    #pragma unroll
    for (int ks = 0; ks < 4; ++ks) An[ks] = A[ks];

    for (int t = t0; t < tEnd; ++t) {
        if (t + 1 < tEnd)
            load_A_tile<STAGED>(h_ptr, src, dst, t + 1, row, hi, E, An);

        // Layer 1: 16 back-to-back WMMAs, all operands in registers.
        v8f acc[4];
        #pragma unroll
        for (int nt = 0; nt < 4; ++nt) {
            v8f c = {};
            #pragma unroll
            for (int ks = 0; ks < 4; ++ks) {
                c = __builtin_amdgcn_wmma_f32_16x16x32_f16(
                        /*neg_a=*/false, A[ks], /*neg_b=*/false, B[nt][ks],
                        /*c_mod=*/(short)0, c, /*reuse_a=*/false, /*reuse_b=*/false);
            }
            acc[nt] = c;
        }

        // Bias + ReLU + layer 2 on C layout:
        // VGPR v, lane L: hidden[m = v + hi*8][n = nt*16 + (L&15)]
        float partial[8];
        #pragma unroll
        for (int v = 0; v < 8; ++v) partial[v] = 0.0f;
        #pragma unroll
        for (int nt = 0; nt < 4; ++nt) {
            #pragma unroll
            for (int v = 0; v < 8; ++v) {
                float hv = acc[nt][v] + b1n[nt];
                hv = hv > 0.0f ? hv : 0.0f;
                partial[v] += hv * w2n[nt];
            }
        }

        // Reduce over the 16 lanes of each half (masks 1,2,4,8 never cross halves).
        #pragma unroll
        for (int m = 1; m < 16; m <<= 1) {
            #pragma unroll
            for (int v = 0; v < 8; ++v)
                partial[v] += __shfl_xor(partial[v], m, 32);
        }

        // lanes 0 and 16 each hold 8 fully-reduced scores (m = v + hi*8).
        if (row == 0) {
            const int base = t * 16 + hi * 8;
            #pragma unroll
            for (int v = 0; v < 8; ++v) {
                const int eo = base + v;
                if (eo < E) out[eo] = partial[v] + b2;
            }
        }

        #pragma unroll
        for (int ks = 0; ks < 4; ++ks) A[ks] = An[ks];
    }
}

// ---------------------------------------------------------------------------
extern "C" void kernel_launch(void* const* d_in, const int* in_sizes, int n_in,
                              void* d_out, int out_size, void* d_ws, size_t ws_size,
                              hipStream_t stream) {
    const float* h   = (const float*)d_in[0];
    const int*   src = (const int*)  d_in[1];
    const int*   dst = (const int*)  d_in[2];
    const float* W1w = (const float*)d_in[3];
    const float* W1b = (const float*)d_in[4];
    const float* W2w = (const float*)d_in[5];
    const float* W2b = (const float*)d_in[6];
    float* out = (float*)d_out;

    const int hElems = in_sizes[0];          // nNodes * 64
    const int E      = in_sizes[1];
    const int nTiles = (E + 15) / 16;
    const int nWaves = (nTiles + TILES_PER_WAVE - 1) / TILES_PER_WAVE;
    const int blocks = (nWaves + 7) / 8;     // 8 waves per 256-thread block

    const size_t need = (size_t)hElems * sizeof(_Float16);
    if (ws_size >= need) {
        _Float16* hf16 = (_Float16*)d_ws;
        const int n4 = hElems / 4;
        cvt_f32_to_f16<<<(n4 + 255) / 256, 256, 0, stream>>>(h, hf16, n4);
        edge_mlp_wmma<true><<<blocks, 256, 0, stream>>>(
            hf16, src, dst, W1w, W1b, W2w, W2b, out, nTiles, E);
    } else {
        edge_mlp_wmma<false><<<blocks, 256, 0, stream>>>(
            h, src, dst, W1w, W1b, W2w, W2b, out, nTiles, E);
    }
}
